// TarsBlock_56212531970331
// MI455X (gfx1250) — matrix operand order repaired and checked by reading, not compiled, not run on hardware
//
#include <hip/hip_runtime.h>
#include <hip/hip_bf16.h>
#include <math.h>

// ---- problem constants (from reference) ----
static constexpr int CB   = 2;     // batch
static constexpr int CL   = 1024;  // seq len
static constexpr int CD   = 1024;  // model dim
static constexpr int CN   = 64;    // state dim
static constexpr int CP   = 64;    // head dim
static constexpr int CH   = 16;    // heads
static constexpr int CE   = 8;     // experts
static constexpr int CR   = 8;     // lora rank
static constexpr int COM  = 32;    // ortho mixer dim
static constexpr int CMEM = 384;   // memory dim
static constexpr int CKC  = 4;     // conv taps

typedef __attribute__((ext_vector_type(16))) _Float16 v16h;
typedef __attribute__((ext_vector_type(8)))  _Float16 v8h;
typedef __attribute__((ext_vector_type(8)))  float    v8f;

// =====================  block reduction helper  =====================
__device__ inline float block_reduce_256(float v, float* red) {
    int tid = threadIdx.x;
    red[tid] = v; __syncthreads();
    for (int off = 128; off > 0; off >>= 1) {
        if (tid < off) red[tid] += red[tid + off];
        __syncthreads();
    }
    float r = red[0]; __syncthreads();
    return r;
}

// async global->LDS 16B copy per lane (CDNA5 ASYNCcnt path)
__device__ inline void async_copy_b128(const float* gsrc, float* lds_dst) {
    unsigned dst = (unsigned)(unsigned long long)lds_dst;   // low 32 bits = LDS offset
    unsigned long long ga = (unsigned long long)gsrc;
    asm volatile("global_load_async_to_lds_b128 %0, %1, off" :: "v"(dst), "v"(ga) : "memory");
}
__device__ inline void async_wait0() {
    asm volatile("s_wait_asynccnt 0x0" ::: "memory");
}

// =====================  f32 -> f16 conversion pass  =====================
__global__ void k_cvt(const float* __restrict__ in, _Float16* __restrict__ out, size_t n) {
    size_t i = (size_t)blockIdx.x * 256 + threadIdx.x;
    if (i < n) out[i] = (_Float16)in[i];
}

// =====================  LayerNorm  =====================
__global__ __launch_bounds__(256) void k_layernorm(const float* __restrict__ x,
                                                   const float* __restrict__ s,
                                                   const float* __restrict__ b,
                                                   float* __restrict__ xn) {
    __shared__ float red[256];
    int row = blockIdx.x;                 // over B*L
    int tid = threadIdx.x;
    const float* xr = x + (size_t)row * CD;
    float ls = 0.f;
    for (int d = tid; d < CD; d += 256) ls += xr[d];
    float mean = block_reduce_256(ls, red) / CD;
    float lv = 0.f;
    for (int d = tid; d < CD; d += 256) { float t = xr[d] - mean; lv += t * t; }
    float var = block_reduce_256(lv, red) / CD;
    float inv = rsqrtf(var + 1e-5f);
    float* o = xn + (size_t)row * CD;
    for (int d = tid; d < CD; d += 256) o[d] = (xr[d] - mean) * inv * s[d] + b[d];
}

// =====================  WMMA f16 GEMM: out = alpha*A@W^T + bias + beta*out ==========
// A: [M,K] f16 row-major, W: [Ndim,K] f16 row-major. Requires K%32==0, Ndim%16==0,
// M%16==0 (true for every launch here). One wave -> 16x64 output (4 WMMA per K-step);
// one block (8 waves) -> 128x64. Fragments are pure contiguous vector loads, no cvt.
__global__ __launch_bounds__(256) void k_gemm(const _Float16* __restrict__ A,
                                              const _Float16* __restrict__ W,
                                              float* __restrict__ out,
                                              int M, int Ndim, int K,
                                              const float* __restrict__ bias,
                                              float alpha, float beta) {
    int wave = threadIdx.x >> 5;
    int lane = threadIdx.x & 31;
    int half = lane >> 4;
    int r    = lane & 15;
    int tileM = blockIdx.x * 128 + wave * 16;
    int tileN = blockIdx.y * 64;
    if (tileM >= M) return;                     // wave-uniform
    int mrow = tileM + r;                       // M%16==0 -> always valid

    bool nvalid[4];
    int  ncl[4];
    #pragma unroll
    for (int nt = 0; nt < 4; nt++) {
        nvalid[nt] = (tileN + nt * 16 + 16) <= Ndim;   // Ndim%16==0 -> whole subtile
        ncl[nt] = nvalid[nt] ? (tileN + nt * 16 + r) : 0;
    }

    v8f acc[4] = {};
    const _Float16* arow = A + (size_t)mrow * K;
    for (int k0 = 0; k0 < K; k0 += 32) {
        __builtin_prefetch(arow + k0 + 128, 0, 1);
        // A fragment: lane holds K = half*16 + [0..15] (contiguous)
        v16h av = *(const v16h*)(arow + k0 + half * 16);
        #pragma unroll
        for (int nt = 0; nt < 4; nt++) {
            // B fragment: lane holds column ncl[nt], K = half*16 + [0..15] (contiguous)
            v16h bv = *(const v16h*)(W + (size_t)ncl[nt] * K + k0 + half * 16);
            acc[nt] = __builtin_amdgcn_wmma_f32_16x16x32_f16(false, av, false, bv,
                                                             (short)0, acc[nt], false, false);
        }
    }
    // store: row = tileM + half*8 + j, col = tileN + nt*16 + (lane%16)
    #pragma unroll
    for (int nt = 0; nt < 4; nt++) {
        if (!nvalid[nt]) continue;
        int n = tileN + nt * 16 + r;
        #pragma unroll
        for (int j = 0; j < 8; j++) {
            int m = tileM + half * 8 + j;
            size_t idx = (size_t)m * Ndim + n;
            float o = alpha * acc[nt][j];
            if (bias) o += bias[n];
            if (beta != 0.f) o += beta * out[idx];
            out[idx] = o;
        }
    }
}

// =====================  depthwise conv + silu, emit conv_new  =====================
__global__ __launch_bounds__(256) void k_conv(const float* __restrict__ zu,
                                              const float* __restrict__ conv_state,
                                              const float* __restrict__ conv_w,
                                              const float* __restrict__ conv_b,
                                              float* __restrict__ uc,
                                              float* __restrict__ o_conv) {
    size_t i = (size_t)blockIdx.x * 256 + threadIdx.x;
    if (i >= (size_t)CB * CL * CD) return;
    int d = i % CD;
    size_t bl = i / CD;
    int l = bl % CL, b = bl / CL;
    float acc = conv_b[d];
    #pragma unroll
    for (int k = 0; k < CKC; k++) {
        int t = l + k;
        float u;
        if (t < CKC) u = conv_state[((size_t)b * CD + d) * CKC + t];
        else         u = zu[((size_t)b * CL + (t - CKC)) * (2 * CD) + CD + d];
        acc += u * conv_w[d * CKC + k];
    }
    uc[i] = acc / (1.f + expf(-acc));       // silu
    if (l < CKC) {                           // conv_new[b][d][l] = u[b][L-KC+l][d]
        o_conv[((size_t)b * CD + d) * CKC + l] =
            zu[((size_t)b * CL + (CL - CKC + l)) * (2 * CD) + CD + d];
    }
}

// =====================  small elementwise kernels  =====================
__global__ void k_softplus(float* buf, int n) {
    int i = blockIdx.x * 256 + threadIdx.x;
    if (i >= n) return;
    float v = buf[i];
    buf[i] = (v > 20.f) ? v : log1pf(expf(v));
}
__global__ void k_expnegexp(float* buf, int n) {
    int i = blockIdx.x * 256 + threadIdx.x;
    if (i >= n) return;
    buf[i] = expf(-expf(buf[i]));
}
__global__ void k_mix(const float* __restrict__ xn, const float* __restrict__ x_prev,
                      float* __restrict__ mix) {
    size_t i = (size_t)blockIdx.x * 256 + threadIdx.x;
    if (i >= (size_t)CB * CL * CD) return;
    int d = i % CD; size_t bl = i / CD; int l = bl % CL, b = bl / CL;
    float shift = (l == 0) ? x_prev[(size_t)b * CD + d]
                           : xn[((size_t)b * CL + l - 1) * CD + d];
    mix[i] = 0.5f * (xn[i] + shift);
}
__global__ void k_core(const float* __restrict__ x, const float* __restrict__ zu,
                       const float* __restrict__ y_ssd, const float* __restrict__ y_rwkv,
                       float* __restrict__ x1) {
    size_t i = (size_t)blockIdx.x * 256 + threadIdx.x;
    if (i >= (size_t)CB * CL * CD) return;
    int d = i % CD; size_t bl = i / CD;
    float z = zu[bl * (2 * CD) + d];
    float sz = z / (1.f + expf(-z));
    x1[i] = x[i] + sz * y_ssd[i] + y_rwkv[i];
}
__global__ void k_badd(float* __restrict__ x1, const float* __restrict__ delta) {
    size_t i = (size_t)blockIdx.x * 256 + threadIdx.x;
    if (i >= (size_t)CB * CL * CD) return;
    int d = i % CD; int b = (i / CD) / CL;
    x1[i] += delta[(size_t)b * CD + d];
}
__global__ void k_axpy(float* __restrict__ dst, const float* __restrict__ src, float a, size_t n) {
    size_t i = (size_t)blockIdx.x * 256 + threadIdx.x;
    if (i < n) dst[i] += a * src[i];
}
__global__ void k_blend(const float* __restrict__ x2, const float* __restrict__ x,
                        const float* __restrict__ nov, float* __restrict__ x3) {
    size_t i = (size_t)blockIdx.x * 256 + threadIdx.x;
    if (i >= (size_t)CB * CL * CD) return;
    int b = (i / CD) / CL;
    float n = nov[b];
    x3[i] = n * x2[i] + (1.f - n) * x[i];
}
__global__ void k_final(const float* __restrict__ x3, const float* __restrict__ scaleb,
                        const float* __restrict__ memproj, float* __restrict__ o_x4) {
    size_t i = (size_t)blockIdx.x * 256 + threadIdx.x;
    if (i >= (size_t)CB * CL * CD) return;
    int d = i % CD; int b = (i / CD) / CL;
    o_x4[i] = x3[i] + scaleb[b] * memproj[(size_t)b * CD + d];
}
__global__ void k_xprev(const float* __restrict__ xn, float* __restrict__ o_xprev) {
    int i = blockIdx.x * 256 + threadIdx.x;
    if (i >= CB * CD) return;
    int b = i / CD, d = i % CD;
    o_xprev[i] = xn[((size_t)b * CL + (CL - 1)) * CD + d];
}
__global__ void k_mean_L(const float* __restrict__ in, float* __restrict__ outm) {
    int i = blockIdx.x * 256 + threadIdx.x;
    if (i >= CB * CD) return;
    int b = i / CD, d = i % CD;
    float s = 0.f;
    for (int l = 0; l < CL; l++) s += in[((size_t)b * CL + l) * CD + d];
    outm[i] = s / CL;
}

// =====================  SSD (Mamba) scan: one block per (b,h)  =====================
__global__ __launch_bounds__(256) void k_ssd_scan(const float* __restrict__ ssd_state,
                                                  const float* __restrict__ dtb,
                                                  const float* __restrict__ A_log,
                                                  const float* __restrict__ Bm,
                                                  const float* __restrict__ Cm,
                                                  const float* __restrict__ uc,
                                                  float* __restrict__ ys,
                                                  float* __restrict__ o_ssd) {
    int bh = blockIdx.x;
    int b = bh / CH, h = bh % CH;
    int tid = threadIdx.x;
    int p = tid & 63, seg = tid >> 6, n0 = seg * 16;
    float s[16];
    const float* s0 = ssd_state + (((size_t)b * CH + h) * CP + p) * CN + n0;
    #pragma unroll
    for (int i = 0; i < 16; i++) s[i] = s0[i];
    float Ah = -expf(A_log[h]);
    __shared__ float Bsh[CN], Csh[CN], red[256];
    for (int t = 0; t < CL; t++) {
        size_t base = (size_t)b * CL + t;
        // async global->LDS staging of the 64-wide B and C rows (wave 0 only)
        if (tid < 32) {
            const float* src = (tid < 16 ? Bm : Cm) + base * CN + (size_t)(tid & 15) * 4;
            float* dst = (tid < 16 ? Bsh : Csh) + (tid & 15) * 4;
            async_copy_b128(src, dst);
        }
        async_wait0();
        __syncthreads();
        float dt_t  = dtb[base * CH + h];
        float u_t   = uc[base * CD + h * CP + p];
        float decay = expf(dt_t * Ah);
        float coef  = dt_t * u_t;
        float partial = 0.f;
        #pragma unroll
        for (int i = 0; i < 16; i++) {
            s[i] = s[i] * decay + coef * Bsh[n0 + i];
            partial += s[i] * Csh[n0 + i];
        }
        red[tid] = partial; __syncthreads();
        if (seg == 0)
            ys[base * CD + h * CP + p] = red[p] + red[p + 64] + red[p + 128] + red[p + 192];
        __syncthreads();
    }
    float* so = o_ssd + (((size_t)b * CH + h) * CP + p) * CN + n0;
    #pragma unroll
    for (int i = 0; i < 16; i++) so[i] = s[i];
}

// =====================  RWKV scan: one block per batch  =====================
__global__ __launch_bounds__(256) void k_rwkv_scan(const float* __restrict__ wkv_state,
                                                   const float* __restrict__ rb,
                                                   const float* __restrict__ kb,
                                                   const float* __restrict__ vb,
                                                   const float* __restrict__ wb,
                                                   float* __restrict__ yr,
                                                   float* __restrict__ o_wkv) {
    int b = blockIdx.x;
    int tid = threadIdx.x;
    int i = tid & 63, seg = tid >> 6, j0 = seg * 16;
    float S[16];
    const float* s0 = wkv_state + ((size_t)b * CN + i) * CN + j0;
    #pragma unroll
    for (int q = 0; q < 16; q++) S[q] = s0[q];
    __shared__ float rsh[CN], ksh[CN], vsh[CN], wsh[CN], red[256];
    for (int t = 0; t < CL; t++) {
        size_t base = ((size_t)b * CL + t) * CN;
        // async global->LDS staging: 64 lanes x 16B cover all four 64-float rows
        if (tid < 64) {
            int which = tid >> 4;      // 0..3 -> r,k,v,w
            int s16   = tid & 15;
            const float* src = (which == 0 ? rb : which == 1 ? kb : which == 2 ? vb : wb)
                               + base + (size_t)s16 * 4;
            float* dst = (which == 0 ? rsh : which == 1 ? ksh : which == 2 ? vsh : wsh)
                         + s16 * 4;
            async_copy_b128(src, dst);
        }
        async_wait0();
        __syncthreads();
        float wi = wsh[i], vi = vsh[i];
        float partial = 0.f;
        #pragma unroll
        for (int q = 0; q < 16; q++) {
            S[q] = S[q] * wi + vi * ksh[j0 + q];
            partial += S[q] * rsh[j0 + q];
        }
        red[tid] = partial; __syncthreads();
        if (seg == 0)
            yr[base + i] = red[i] + red[i + 64] + red[i + 128] + red[i + 192];
        __syncthreads();
    }
    float* so = o_wkv + ((size_t)b * CN + i) * CN + j0;
    #pragma unroll
    for (int q = 0; q < 16; q++) so[q] = S[q];
}

// =====================  RAG: delta = 0.1 * (rag_state @ (x1mean @ Wq^T)) @ Wo^T ======
__global__ __launch_bounds__(256) void k_rag(const float* __restrict__ x1mean,
                                             const float* __restrict__ rag_state,
                                             const float* __restrict__ W_ragq,
                                             const float* __restrict__ W_rago,
                                             float* __restrict__ delta) {
    int b = blockIdx.x, tid = threadIdx.x;
    __shared__ float q[CN], info[CN];
    if (tid < CN) {
        float s = 0.f;
        for (int d = 0; d < CD; d++) s += x1mean[(size_t)b * CD + d] * W_ragq[(size_t)tid * CD + d];
        q[tid] = s;
    }
    __syncthreads();
    if (tid < CN) {
        float s = 0.f;
        for (int j = 0; j < CN; j++) s += rag_state[((size_t)b * CN + tid) * CN + j] * q[j];
        info[tid] = s;
    }
    __syncthreads();
    for (int d = tid; d < CD; d += 256) {
        float s = 0.f;
        for (int n = 0; n < CN; n++) s += info[n] * W_rago[(size_t)d * CN + n];
        delta[(size_t)b * CD + d] = 0.1f * s;
    }
}

// =====================  Cayley: solve (I-Ask) Qc = (I+Ask), 32x32 Gauss-Jordan =======
__global__ void k_cayley(const float* __restrict__ W_skew, float* __restrict__ Qc) {
    __shared__ float Mrow[COM][2 * COM];
    int tid = threadIdx.x;
    if (tid < COM) {
        for (int j = 0; j < COM; j++) {
            float a = 0.5f * (W_skew[tid * COM + j] - W_skew[j * COM + tid]);
            Mrow[tid][j]       = (tid == j ? 1.f : 0.f) - a;
            Mrow[tid][COM + j] = (tid == j ? 1.f : 0.f) + a;
        }
    }
    __syncthreads();
    for (int c = 0; c < COM; c++) {
        if (tid == c) {
            float piv = Mrow[c][c];
            for (int j = 0; j < 2 * COM; j++) Mrow[c][j] /= piv;
        }
        __syncthreads();
        if (tid < COM && tid != c) {
            float f = Mrow[tid][c];
            for (int j = 0; j < 2 * COM; j++) Mrow[tid][j] -= f * Mrow[c][j];
        }
        __syncthreads();
    }
    if (tid < COM)
        for (int j = 0; j < COM; j++) Qc[tid * COM + j] = Mrow[tid][COM + j];
}

// =====================  MoE router + top-2 LoRA (per token)  =====================
__global__ __launch_bounds__(256) void k_moe(const float* __restrict__ x1b,
                                             const float* __restrict__ W_router,
                                             const float* __restrict__ lora_A,
                                             const float* __restrict__ lora_B,
                                             float* __restrict__ x2,
                                             float* __restrict__ accums) {
    int row = blockIdx.x, tid = threadIdx.x;
    const float* xr = x1b + (size_t)row * CD;
    __shared__ float red[256];
    __shared__ float probs[CE];
    __shared__ int   top[2];
    __shared__ float gateSh[2];
    __shared__ float down[2][CR];
    for (int e = 0; e < CE; e++) {
        float p = 0.f;
        for (int d = tid; d < CD; d += 256) p += xr[d] * W_router[(size_t)e * CD + d];
        float s = block_reduce_256(p, red);
        if (tid == 0) probs[e] = s;
        __syncthreads();
    }
    if (tid == 0) {
        float mx = -1e30f;
        for (int e = 0; e < CE; e++) mx = fmaxf(mx, probs[e]);
        float s = 0.f;
        for (int e = 0; e < CE; e++) { probs[e] = expf(probs[e] - mx); s += probs[e]; }
        for (int e = 0; e < CE; e++) probs[e] /= s;
        int i0 = 0;
        for (int e = 1; e < CE; e++) if (probs[e] > probs[i0]) i0 = e;
        int i1 = (i0 == 0) ? 1 : 0;
        for (int e = 0; e < CE; e++) if (e != i0 && probs[e] > probs[i1]) i1 = e;
        top[0] = i0; top[1] = i1;
        float gs = probs[i0] + probs[i1] + 1e-9f;
        gateSh[0] = probs[i0] / gs; gateSh[1] = probs[i1] / gs;
        for (int e = 0; e < CE; e++) atomicAdd(&accums[e], probs[e]);
        atomicAdd(&accums[CE + i0], 1.f);
        atomicAdd(&accums[CE + i1], 1.f);
    }
    __syncthreads();
    for (int s2 = 0; s2 < 2; s2++) {
        int e = top[s2];
        for (int r2 = 0; r2 < CR; r2++) {
            float p = 0.f;
            for (int d = tid; d < CD; d += 256)
                p += xr[d] * lora_A[((size_t)e * CD + d) * CR + r2];
            float s = block_reduce_256(p, red);
            if (tid == 0) down[s2][r2] = s * gateSh[s2];
            __syncthreads();
        }
    }
    for (int d = tid; d < CD; d += 256) {
        float acc = xr[d];
        for (int s2 = 0; s2 < 2; s2++) {
            int e = top[s2];
            #pragma unroll
            for (int r2 = 0; r2 < CR; r2++)
                acc += down[s2][r2] * lora_B[((size_t)e * CR + r2) * CD + d];
        }
        x2[(size_t)row * CD + d] = acc;
    }
}

__global__ void k_aux(const float* __restrict__ accums, float* __restrict__ o_aux) {
    if (threadIdx.x == 0 && blockIdx.x == 0) {
        float inv = 1.f / (float)(CB * CL);
        float s = 0.f;
        for (int e = 0; e < CE; e++) s += (accums[e] * inv) * (accums[CE + e] * inv);
        o_aux[0] = CE * s;
    }
}

// =====================  novelty gate  =====================
__global__ __launch_bounds__(256) void k_novelty(const float* __restrict__ h_old,
                                                 const float* __restrict__ h_new,
                                                 const float* __restrict__ W_nov,
                                                 const float* __restrict__ b_nov,
                                                 float* __restrict__ nov) {
    int b = blockIdx.x, tid = threadIdx.x;
    __shared__ float red[256];
    float p = 0.f;
    for (int d = tid; d < CD; d += 256)
        p += h_old[(size_t)b * CD + d] * W_nov[d] + h_new[(size_t)b * CD + d] * W_nov[CD + d];
    float s = block_reduce_256(p, red);
    if (tid == 0) nov[b] = 1.f / (1.f + expf(-(s + b_nov[0])));
}

// =====================  memory sim*gate scale  =====================
__global__ __launch_bounds__(256) void k_memory(const float* __restrict__ h_post,
                                                const float* __restrict__ memory_vec,
                                                const float* __restrict__ W_mq,
                                                const float* __restrict__ W_mg,
                                                const float* __restrict__ b_mg,
                                                float* __restrict__ scaleb) {
    int b = blockIdx.x, tid = threadIdx.x;
    __shared__ float red[256];
    float dotv = 0.f, nq = 0.f, nm = 0.f, gacc = 0.f;
    for (int j = tid; j < CMEM; j += 256) {
        float hq = 0.f;
        for (int d = 0; d < CD; d++) hq += h_post[(size_t)b * CD + d] * W_mq[(size_t)j * CD + d];
        float mv = memory_vec[(size_t)b * CMEM + j];
        dotv += hq * mv; nq += hq * hq; nm += mv * mv;
        gacc += mv * W_mg[CD + j];
    }
    for (int d = tid; d < CD; d += 256) gacc += h_post[(size_t)b * CD + d] * W_mg[d];
    float rdot = block_reduce_256(dotv, red);
    float rnq  = block_reduce_256(nq, red);
    float rnm  = block_reduce_256(nm, red);
    float rg   = block_reduce_256(gacc, red);
    if (tid == 0) {
        float sim = rdot / (sqrtf(rnq) * sqrtf(rnm) + 1e-8f);
        float gm  = 1.f / (1.f + expf(-(rg + b_mg[0])));
        scaleb[b] = sim * gm;
    }
}
__global__ void k_memproj(const float* __restrict__ memory_vec, const float* __restrict__ W_mp,
                          float* __restrict__ memproj) {
    int i = blockIdx.x * 256 + threadIdx.x;
    if (i >= CB * CD) return;
    int b = i / CD, d = i % CD;
    float s = 0.f;
    for (int j = 0; j < CMEM; j++) s += memory_vec[(size_t)b * CMEM + j] * W_mp[(size_t)d * CMEM + j];
    memproj[i] = s;
}

// ==================================================================
extern "C" void kernel_launch(void* const* d_in, const int* in_sizes, int n_in,
                              void* d_out, int out_size, void* d_ws, size_t ws_size,
                              hipStream_t stream) {
    const float* x          = (const float*)d_in[0];
    const float* wkv_state  = (const float*)d_in[1];
    const float* x_prev     = (const float*)d_in[2];
    const float* memory_vec = (const float*)d_in[3];
    const float* rag_state  = (const float*)d_in[4];
    const float* ssd_state  = (const float*)d_in[5];
    const float* conv_state = (const float*)d_in[6];
    const float* norm_scale = (const float*)d_in[7];
    const float* norm_bias  = (const float*)d_in[8];
    const float* W_in       = (const float*)d_in[9];
    const float* conv_w     = (const float*)d_in[10];
    const float* conv_b     = (const float*)d_in[11];
    const float* W_dt       = (const float*)d_in[12];
    const float* dt_bias    = (const float*)d_in[13];
    const float* A_log      = (const float*)d_in[14];
    const float* W_B        = (const float*)d_in[15];
    const float* W_C        = (const float*)d_in[16];
    const float* W_ssd_out  = (const float*)d_in[17];
    const float* W_r        = (const float*)d_in[18];
    const float* W_k        = (const float*)d_in[19];
    const float* W_v        = (const float*)d_in[20];
    const float* W_w        = (const float*)d_in[21];
    const float* W_rwkv_out = (const float*)d_in[22];
    const float* W_skew     = (const float*)d_in[23];
    const float* W_om_in    = (const float*)d_in[24];
    const float* W_om_out   = (const float*)d_in[25];
    const float* W_router   = (const float*)d_in[26];
    const float* lora_A     = (const float*)d_in[27];
    const float* lora_B     = (const float*)d_in[28];
    const float* W_nov      = (const float*)d_in[29];
    const float* b_nov      = (const float*)d_in[30];
    const float* W_ragq     = (const float*)d_in[31];
    const float* W_rago     = (const float*)d_in[32];
    const float* W_mq       = (const float*)d_in[33];
    const float* W_mp       = (const float*)d_in[34];
    const float* W_mg       = (const float*)d_in[35];
    const float* b_mg       = (const float*)d_in[36];

    float* out = (float*)d_out;
    float* o_x4    = out;
    float* o_wkv   = o_x4 + (size_t)CB * CL * CD;
    float* o_xprev = o_wkv + (size_t)CB * CN * CN;
    float* o_ssd   = o_xprev + (size_t)CB * CD;
    float* o_conv  = o_ssd + (size_t)CB * CH * CP * CN;
    float* o_aux   = o_conv + (size_t)CB * CD * CKC;

    // workspace carve-out (all chunk sizes are multiples of 16 floats until the tail)
    float* w = (float*)d_ws;
    auto take   = [&](size_t n) { float* p = w; w += n; return p; };
    auto take16 = [&](size_t n) { _Float16* p = (_Float16*)w; w += (n + 1) / 2; return p; };
    const size_t BLD = (size_t)CB * CL * CD;
    const size_t BLN = (size_t)CB * CL * CN;
    float* xn      = take(BLD);
    float* zu      = take(2 * BLD);
    float* uc      = take(BLD);
    float* dtb     = take((size_t)CB * CL * CH);
    float* Bm      = take(BLN);
    float* Cm      = take(BLN);
    float* ys      = take(BLD);
    float* y_ssd   = take(BLD);
    float* mixb    = take(BLD);
    float* rb      = take(BLN);
    float* kb      = take(BLN);
    float* vb      = take(BLN);
    float* wb      = take(BLN);
    float* yr      = take(BLN);
    float* y_rwkv  = take(BLD);
    float* x1      = take(BLD);
    float* t1      = take((size_t)CB * CL * COM);
    float* t2      = take((size_t)CB * CL * COM);
    float* x1b     = take(BLD);
    float* x2      = take(BLD);
    float* x3      = take(BLD);
    float* Qc      = take(COM * COM);
    // f16 activation copies
    _Float16* xn16   = take16(BLD);
    _Float16* ys16   = take16(BLD);
    _Float16* mixb16 = take16(BLD);
    _Float16* yr16   = take16(BLN);
    _Float16* x1_16  = take16(BLD);
    _Float16* t1_16  = take16((size_t)CB * CL * COM);
    _Float16* t2_16  = take16((size_t)CB * CL * COM);
    _Float16* Qc16   = take16(COM * COM);
    // f16 weight copies
    _Float16* W_in16   = take16((size_t)2 * CD * CD);
    _Float16* W_dt16   = take16((size_t)CH * CD);
    _Float16* W_B16    = take16((size_t)CN * CD);
    _Float16* W_C16    = take16((size_t)CN * CD);
    _Float16* W_so16   = take16((size_t)CD * CD);
    _Float16* W_r16    = take16((size_t)CN * CD);
    _Float16* W_k16    = take16((size_t)CN * CD);
    _Float16* W_v16    = take16((size_t)CN * CD);
    _Float16* W_w16    = take16((size_t)CN * CD);
    _Float16* W_ro16   = take16((size_t)CD * CN);
    _Float16* W_omi16  = take16((size_t)COM * CD);
    _Float16* W_omo16  = take16((size_t)CD * COM);
    // small f32 tail buffers
    float* x1mean  = take(CB * CD);
    float* deltar  = take(CB * CD);
    float* hold    = take(CB * CD);
    float* hnew    = take(CB * CD);
    float* hpost   = take(CB * CD);
    float* memproj = take(CB * CD);
    float* nov     = take(CB);
    float* scaleb  = take(CB);
    float* accums  = take(2 * CE);

    const int M = CB * CL; // 2048 token rows
    dim3 blk(256);
    auto gemmGrid = [](int m, int n) { return dim3((m + 127) / 128, (n + 63) / 64); };
    auto cvtGrid  = [](size_t n) { return dim3((unsigned)((n + 255) / 256)); };
    int flatBLD = (int)((BLD + 255) / 256);
    int flatBLN = (int)((BLN + 255) / 256);

    // 0) one-time (per launch) weight f32->f16 conversions
    k_cvt<<<cvtGrid(2 * CD * CD), blk, 0, stream>>>(W_in, W_in16, (size_t)2 * CD * CD);
    k_cvt<<<cvtGrid(CH * CD), blk, 0, stream>>>(W_dt, W_dt16, (size_t)CH * CD);
    k_cvt<<<cvtGrid(CN * CD), blk, 0, stream>>>(W_B, W_B16, (size_t)CN * CD);
    k_cvt<<<cvtGrid(CN * CD), blk, 0, stream>>>(W_C, W_C16, (size_t)CN * CD);
    k_cvt<<<cvtGrid(CD * CD), blk, 0, stream>>>(W_ssd_out, W_so16, (size_t)CD * CD);
    k_cvt<<<cvtGrid(CN * CD), blk, 0, stream>>>(W_r, W_r16, (size_t)CN * CD);
    k_cvt<<<cvtGrid(CN * CD), blk, 0, stream>>>(W_k, W_k16, (size_t)CN * CD);
    k_cvt<<<cvtGrid(CN * CD), blk, 0, stream>>>(W_v, W_v16, (size_t)CN * CD);
    k_cvt<<<cvtGrid(CN * CD), blk, 0, stream>>>(W_w, W_w16, (size_t)CN * CD);
    k_cvt<<<cvtGrid(CD * CN), blk, 0, stream>>>(W_rwkv_out, W_ro16, (size_t)CD * CN);
    k_cvt<<<cvtGrid(COM * CD), blk, 0, stream>>>(W_om_in, W_omi16, (size_t)COM * CD);
    k_cvt<<<cvtGrid(CD * COM), blk, 0, stream>>>(W_om_out, W_omo16, (size_t)CD * COM);

    // 1) LayerNorm (+ f16 copy of xn)
    k_layernorm<<<M, blk, 0, stream>>>(x, norm_scale, norm_bias, xn);
    k_cvt<<<cvtGrid(BLD), blk, 0, stream>>>(xn, xn16, BLD);
    // 2) zu = xn @ W_in^T   (2048x2048x1024 WMMA GEMM)
    k_gemm<<<gemmGrid(M, 2 * CD), blk, 0, stream>>>(xn16, W_in16, zu, M, 2 * CD, CD, nullptr, 1.f, 0.f);
    // 3) conv + silu (+ conv_new output)
    k_conv<<<flatBLD, blk, 0, stream>>>(zu, conv_state, conv_w, conv_b, uc, o_conv);
    // 4) dt = softplus(xn @ W_dt^T + dt_bias)
    k_gemm<<<gemmGrid(M, CH), blk, 0, stream>>>(xn16, W_dt16, dtb, M, CH, CD, dt_bias, 1.f, 0.f);
    k_softplus<<<(M * CH + 255) / 256, blk, 0, stream>>>(dtb, M * CH);
    // 5) B, C projections
    k_gemm<<<gemmGrid(M, CN), blk, 0, stream>>>(xn16, W_B16, Bm, M, CN, CD, nullptr, 1.f, 0.f);
    k_gemm<<<gemmGrid(M, CN), blk, 0, stream>>>(xn16, W_C16, Cm, M, CN, CD, nullptr, 1.f, 0.f);
    // 6) SSD scan (+ ssd_new output)
    k_ssd_scan<<<CB * CH, blk, 0, stream>>>(ssd_state, dtb, A_log, Bm, Cm, uc, ys, o_ssd);
    // 7) y_ssd = ys @ W_ssd_out^T
    k_cvt<<<cvtGrid(BLD), blk, 0, stream>>>(ys, ys16, BLD);
    k_gemm<<<gemmGrid(M, CD), blk, 0, stream>>>(ys16, W_so16, y_ssd, M, CD, CD, nullptr, 1.f, 0.f);
    // 8) RWKV branch
    k_mix<<<flatBLD, blk, 0, stream>>>(xn, x_prev, mixb);
    k_cvt<<<cvtGrid(BLD), blk, 0, stream>>>(mixb, mixb16, BLD);
    k_gemm<<<gemmGrid(M, CN), blk, 0, stream>>>(mixb16, W_r16, rb, M, CN, CD, nullptr, 1.f, 0.f);
    k_gemm<<<gemmGrid(M, CN), blk, 0, stream>>>(mixb16, W_k16, kb, M, CN, CD, nullptr, 1.f, 0.f);
    k_gemm<<<gemmGrid(M, CN), blk, 0, stream>>>(mixb16, W_v16, vb, M, CN, CD, nullptr, 1.f, 0.f);
    k_gemm<<<gemmGrid(M, CN), blk, 0, stream>>>(mixb16, W_w16, wb, M, CN, CD, nullptr, 1.f, 0.f);
    k_expnegexp<<<flatBLN, blk, 0, stream>>>(wb, (int)BLN);
    k_rwkv_scan<<<CB, blk, 0, stream>>>(wkv_state, rb, kb, vb, wb, yr, o_wkv);
    k_cvt<<<cvtGrid(BLN), blk, 0, stream>>>(yr, yr16, BLN);
    k_gemm<<<gemmGrid(M, CD), blk, 0, stream>>>(yr16, W_ro16, y_rwkv, M, CD, CN, nullptr, 1.f, 0.f);
    // 9) core combine: x1 = x + silu(z)*y_ssd + y_rwkv
    k_core<<<flatBLD, blk, 0, stream>>>(x, zu, y_ssd, y_rwkv, x1);
    // 10) RAG broadcast add
    k_mean_L<<<(CB * CD + 255) / 256, blk, 0, stream>>>(x1, x1mean);
    k_rag<<<CB, blk, 0, stream>>>(x1mean, rag_state, W_ragq, W_rago, deltar);
    k_badd<<<flatBLD, blk, 0, stream>>>(x1, deltar);
    // 11) Cayley orthogonal mixer: x1b = x1 + x1 @ Wom_in^T @ Qc^T @ Wom_out^T
    k_cayley<<<1, 32, 0, stream>>>(W_skew, Qc);
    k_cvt<<<cvtGrid(COM * COM), blk, 0, stream>>>(Qc, Qc16, (size_t)COM * COM);
    k_cvt<<<cvtGrid(BLD), blk, 0, stream>>>(x1, x1_16, BLD);
    k_gemm<<<gemmGrid(M, COM), blk, 0, stream>>>(x1_16, W_omi16, t1, M, COM, CD, nullptr, 1.f, 0.f);
    k_cvt<<<cvtGrid((size_t)M * COM), blk, 0, stream>>>(t1, t1_16, (size_t)M * COM);
    k_gemm<<<gemmGrid(M, COM), blk, 0, stream>>>(t1_16, Qc16, t2, M, COM, COM, nullptr, 1.f, 0.f);
    k_cvt<<<cvtGrid((size_t)M * COM), blk, 0, stream>>>(t2, t2_16, (size_t)M * COM);
    k_gemm<<<gemmGrid(M, CD), blk, 0, stream>>>(t2_16, W_omo16, x1b, M, CD, COM, nullptr, 1.f, 0.f);
    k_axpy<<<flatBLD, blk, 0, stream>>>(x1b, x1, 1.f, BLD);
    // 12) MoE router + top-2 LoRA, aux stats
    hipMemsetAsync(accums, 0, 2 * CE * sizeof(float), stream);
    k_moe<<<M, blk, 0, stream>>>(x1b, W_router, lora_A, lora_B, x2, accums);
    k_aux<<<1, 32, 0, stream>>>(accums, o_aux);
    // 13) novelty blend
    k_mean_L<<<(CB * CD + 255) / 256, blk, 0, stream>>>(x, hold);
    k_mean_L<<<(CB * CD + 255) / 256, blk, 0, stream>>>(x2, hnew);
    k_novelty<<<CB, blk, 0, stream>>>(hold, hnew, W_nov, b_nov, nov);
    k_blend<<<flatBLD, blk, 0, stream>>>(x2, x, nov, x3);
    // 14) memory gating -> x4 (output)
    k_mean_L<<<(CB * CD + 255) / 256, blk, 0, stream>>>(x3, hpost);
    k_memory<<<CB, blk, 0, stream>>>(hpost, memory_vec, W_mq, W_mg, b_mg, scaleb);
    k_memproj<<<(CB * CD + 255) / 256, blk, 0, stream>>>(memory_vec, W_mp, memproj);
    k_final<<<flatBLD, blk, 0, stream>>>(x3, scaleb, memproj, o_x4);
    // 15) x_prev_new output
    k_xprev<<<(CB * CD + 255) / 256, blk, 0, stream>>>(xn, o_xprev);
}
